// HORN_SL_22393959482154
// MI455X (gfx1250) — matrix operand order repaired and checked by reading, not compiled, not run on hardware
//
#include <hip/hip_runtime.h>
#include <hip/hip_bf16.h>
#include <math.h>

// ---------------- problem constants ----------------
#define T_STEPS 256
#define B_SZ    256
#define I_SZ    128
#define N_SZ    1024
#define O_SZ    64

#define H_DT     0.05f
#define ALPHA_C  1.0f
#define OMEGA_C  6.2831853071795864769f   // 2*pi*1Hz
#define GAMMA_C  0.1f
#define LAM_C    1.0f
#define GAINREC  0.03125f                  // 1/sqrt(1024)

typedef __attribute__((ext_vector_type(16))) __bf16 v16bf;
typedef __attribute__((ext_vector_type(8)))  __bf16 v8bf;
typedef __attribute__((ext_vector_type(8)))  float  v8f;

// ---------------- one-time prep kernels ----------------
__global__ __launch_bounds__(256) void hsl_cvt_w_bf16(const float* __restrict__ W,
                                                      __bf16* __restrict__ Wb, int n) {
    int i = blockIdx.x * 256 + threadIdx.x;
    if (i < n) Wb[i] = (__bf16)W[i];
}

__global__ __launch_bounds__(256) void hsl_forcing_t0(const float* __restrict__ batch0,
                                                      const float* __restrict__ W_ih,
                                                      const float* __restrict__ b_ih,
                                                      float* __restrict__ F) {
    int tid = blockIdx.x * 256 + threadIdx.x;       // one per (b, n)
    int b = tid >> 10;
    int n = tid & (N_SZ - 1);
    float s = b_ih[n];
    const float* bi = batch0 + b * I_SZ;
    const float* wi = W_ih  + n * I_SZ;
    #pragma unroll 4
    for (int k = 0; k < I_SZ; ++k) s = fmaf(bi[k], wi[k], s);
    F[tid] = ALPHA_C * tanhf(s);
}

__global__ __launch_bounds__(256) void hsl_forcing_const(const float* __restrict__ b_ih,
                                                         float* __restrict__ Fc) {
    int n = blockIdx.x * 256 + threadIdx.x;
    if (n < N_SZ) Fc[n] = ALPHA_C * tanhf(b_ih[n]);
}

__global__ __launch_bounds__(256) void hsl_zero(float* __restrict__ p, int n) {
    int i = blockIdx.x * 256 + threadIdx.x;
    if (i < n) p[i] = 0.0f;
}

// final projection: out(b,o) = x_T(b,:) . W_ho(o,:) + b_ho(o)
__global__ __launch_bounds__(256) void hsl_out_proj(const float* __restrict__ xT,
                                                    const float* __restrict__ W_ho,
                                                    const float* __restrict__ b_ho,
                                                    float* __restrict__ out) {
    int tid = blockIdx.x * 256 + threadIdx.x;       // one per (b, o)
    int b = tid >> 6;
    int o = tid & (O_SZ - 1);
    float s = b_ho[o];
    const float* xr = xT   + b * N_SZ;
    const float* wr = W_ho + o * N_SZ;
    #pragma unroll 4
    for (int k = 0; k < N_SZ; ++k) s = fmaf(xr[k], wr[k], s);
    out[tid] = s;
}

// ---------------- RK4 stage kernel (WMMA, LDS-free) ----------------
// A operands (sxb/syb, bf16, B x N) were written by the previous stage/init.
// GEMM: s_x @ W_hh.T and s_y @ W_hh.T via v_wmma_f32_16x16x32_bf16; fragments
// are loaded straight from global (L2-resident) -- per-lane chunks hold
// k = 16*(lane/16) + s (same order for A and B), contiguous 16B/32B per lane.
// Epilogue computes Stuart-Landau dynamics in f32 and emits the NEXT stage's
// bf16 A operand (x + cNext*k), so conversion never touches the inner loop.
// FOLD=true (stage 4): writes x_new/y_new (f32 ping-pong) and its bf16 copy.
template <bool FOLD>
__global__ __launch_bounds__(256)
void hsl_stage(const __bf16* __restrict__ sxb, const __bf16* __restrict__ syb,
               const float* __restrict__ x,   const float* __restrict__ y,
               const float* __restrict__ kxp, const float* __restrict__ kyp,
               float cC, float cNext,
               const float* __restrict__ F, int fStride,      // N or 0 (broadcast)
               const __bf16* __restrict__ Wbf,
               const float* __restrict__ b_hh,
               float* __restrict__ outX, float* __restrict__ outY,
               __bf16* __restrict__ nxtXb, __bf16* __restrict__ nxtYb,
               const float* __restrict__ k1x, const float* __restrict__ k1y,
               const float* __restrict__ k2x, const float* __restrict__ k2y,
               const float* __restrict__ k3x, const float* __restrict__ k3y) {
    const int lane = threadIdx.x & 31;
    const int wave = threadIdx.x >> 5;
    const int wm   = wave >> 2;                       // 2 waves along M
    const int wn   = wave & 3;                        // 4 waves along N
    const int m0   = blockIdx.y * 32 + wm * 16;       // C row base (batch)
    const int n0   = blockIdx.x * 64 + wn * 16;       // C col base (hidden)

    // A fragment (16x32 bf16): all lanes row M=lane&15; lanes>=16 take K=16..31
    const int arow = m0 + (lane & 15);
    const int kpa  = (lane >> 4) << 4;
    // B fragment (32x16 bf16): column N=lane&15; lanes>=16 take K=16..31
    const int brow = n0 + (lane & 15);
    const int kpb  = (lane >> 4) << 4;

    const __bf16* axp = sxb + arow * N_SZ + kpa;
    const __bf16* ayp = syb + arow * N_SZ + kpa;
    const __bf16* bwp = Wbf + brow * N_SZ + kpb;

    v8f accx = {};
    v8f accy = {};

    #pragma unroll 4
    for (int k0 = 0; k0 < N_SZ; k0 += 32) {
        v8bf axlo = *(const v8bf*)(axp + k0);
        v8bf axhi = *(const v8bf*)(axp + k0 + 8);
        v8bf aylo = *(const v8bf*)(ayp + k0);
        v8bf ayhi = *(const v8bf*)(ayp + k0 + 8);
        v16bf bw  = *(const v16bf*)(bwp + k0);
        v16bf ax  = __builtin_shufflevector(axlo, axhi, 0,1,2,3,4,5,6,7,8,9,10,11,12,13,14,15);
        v16bf ay  = __builtin_shufflevector(aylo, ayhi, 0,1,2,3,4,5,6,7,8,9,10,11,12,13,14,15);
        accx = __builtin_amdgcn_wmma_f32_16x16x32_bf16(false, ax, false, bw,
                                                       (short)0, accx, false, false);
        accy = __builtin_amdgcn_wmma_f32_16x16x32_bf16(false, ay, false, bw,
                                                       (short)0, accy, false, false);
    }

    // --- epilogue: Stuart-Landau dynamics per C element ---
    // C layout: lanes 0-15 N=lane, VGPR i -> M=i ; lanes 16-31 N=lane-16, M=8+i
    const int nn    = n0 + (lane & 15);
    const int bbase = m0 + ((lane >> 4) << 3);
    const float bh  = b_hh[nn];
    #pragma unroll
    for (int i = 0; i < 8; ++i) {
        const int b   = bbase + i;
        const int idx = b * N_SZ + nn;
        float xv = x[idx], yv = y[idx];
        float sx = fmaf(cC, kxp[idx], xv);
        float sy = fmaf(cC, kyp[idx], yv);
        float r2 = fmaf(sx, sx, sy * sy);
        float a  = (LAM_C - GAMMA_C) - r2;
        float fz = F[b * fStride + nn];
        float dx = a * sx - OMEGA_C * sy + GAINREC * (accx[i] + bh) + fz;
        float dy = a * sy + OMEGA_C * sx + GAINREC * (accy[i] + bh);
        if (FOLD) {
            float xn = xv + (H_DT / 6.0f) * (k1x[idx] + 2.0f * k2x[idx] + 2.0f * k3x[idx] + dx);
            float yn = yv + (H_DT / 6.0f) * (k1y[idx] + 2.0f * k2y[idx] + 2.0f * k3y[idx] + dy);
            outX[idx]  = xn;
            outY[idx]  = yn;
            nxtXb[idx] = (__bf16)xn;          // A for next timestep's stage 1
            nxtYb[idx] = (__bf16)yn;
        } else {
            outX[idx]  = dx;
            outY[idx]  = dy;
            nxtXb[idx] = (__bf16)fmaf(cNext, dx, xv);  // A for next RK4 stage
            nxtYb[idx] = (__bf16)fmaf(cNext, dy, yv);
        }
    }
}

// ---------------- host driver ----------------
extern "C" void kernel_launch(void* const* d_in, const int* in_sizes, int n_in,
                              void* d_out, int out_size, void* d_ws, size_t ws_size,
                              hipStream_t stream) {
    const float* batch = (const float*)d_in[0];   // (T,B,I) ; t=0 slice is first B*I
    const float* W_ih  = (const float*)d_in[1];   // (N,I)
    const float* b_ih  = (const float*)d_in[2];   // (N)
    const float* W_hh  = (const float*)d_in[3];   // (N,N)
    const float* b_hh  = (const float*)d_in[4];   // (N)
    const float* W_ho  = (const float*)d_in[5];   // (O,N)
    const float* b_ho  = (const float*)d_in[6];   // (O)

    char* ws = (char*)d_ws;
    const size_t MB = (size_t)1 << 20;
    const size_t HB = (size_t)512 << 10;          // bf16 state buffer: 512 KB
    __bf16* Wbf = (__bf16*)(ws + 0 * MB);         // 2 MB
    float*  Ft0 = (float*)(ws + 2 * MB);          // 1 MB  (B x N)
    float*  Fc  = (float*)(ws + 3 * MB);          // 4 KB  (N)
    float*  xA  = (float*)(ws + 4 * MB);
    float*  yA  = (float*)(ws + 5 * MB);
    float*  xB  = (float*)(ws + 6 * MB);
    float*  yB  = (float*)(ws + 7 * MB);
    float*  k1x = (float*)(ws + 8 * MB);
    float*  k1y = (float*)(ws + 9 * MB);
    float*  k2x = (float*)(ws + 10 * MB);
    float*  k2y = (float*)(ws + 11 * MB);
    float*  k3x = (float*)(ws + 12 * MB);
    float*  k3y = (float*)(ws + 13 * MB);
    __bf16* xb   = (__bf16*)(ws + 14 * MB + 0 * HB);  // stage-1 A (bf16 of x)
    __bf16* yb   = (__bf16*)(ws + 14 * MB + 1 * HB);
    __bf16* s2xb = (__bf16*)(ws + 14 * MB + 2 * HB);  // stage-2 A
    __bf16* s2yb = (__bf16*)(ws + 14 * MB + 3 * HB);
    __bf16* s3xb = (__bf16*)(ws + 14 * MB + 4 * HB);  // stage-3 A
    __bf16* s3yb = (__bf16*)(ws + 14 * MB + 5 * HB);
    __bf16* s4xb = (__bf16*)(ws + 14 * MB + 6 * HB);  // stage-4 A
    __bf16* s4yb = (__bf16*)(ws + 14 * MB + 7 * HB);

    // one-time prep
    hsl_cvt_w_bf16<<<(N_SZ * N_SZ) / 256, 256, 0, stream>>>(W_hh, Wbf, N_SZ * N_SZ);
    hsl_forcing_t0<<<(B_SZ * N_SZ) / 256, 256, 0, stream>>>(batch, W_ih, b_ih, Ft0);
    hsl_forcing_const<<<(N_SZ + 255) / 256, 256, 0, stream>>>(b_ih, Fc);
    hsl_zero<<<(B_SZ * N_SZ) / 256, 256, 0, stream>>>(xA, B_SZ * N_SZ);
    hsl_zero<<<(B_SZ * N_SZ) / 256, 256, 0, stream>>>(yA, B_SZ * N_SZ);
    hsl_zero<<<(B_SZ * N_SZ / 2) / 256, 256, 0, stream>>>((float*)xb, B_SZ * N_SZ / 2);
    hsl_zero<<<(B_SZ * N_SZ / 2) / 256, 256, 0, stream>>>((float*)yb, B_SZ * N_SZ / 2);

    dim3 grid(N_SZ / 64, B_SZ / 32);   // (16, 8) blocks, 8 waves each => 1024 wave tiles
    dim3 blk(256);

    float* xc = xA; float* yc = yA;
    float* xn = xB; float* yn = yB;
    for (int ts = 0; ts < T_STEPS; ++ts) {
        const float* F = (ts == 0) ? Ft0 : Fc;
        const int   fs = (ts == 0) ? N_SZ : 0;
        // k1 = f(x,y): A = bf16(x); emits stage-2 A = bf16(x + H/2*k1)
        hsl_stage<false><<<grid, blk, 0, stream>>>(xb, yb, xc, yc, xc, yc,
                                                   0.0f, 0.5f * H_DT, F, fs,
                                                   Wbf, b_hh, k1x, k1y, s2xb, s2yb,
                                                   nullptr, nullptr, nullptr, nullptr, nullptr, nullptr);
        // k2: emits stage-3 A = bf16(x + H/2*k2)
        hsl_stage<false><<<grid, blk, 0, stream>>>(s2xb, s2yb, xc, yc, k1x, k1y,
                                                   0.5f * H_DT, 0.5f * H_DT, F, fs,
                                                   Wbf, b_hh, k2x, k2y, s3xb, s3yb,
                                                   nullptr, nullptr, nullptr, nullptr, nullptr, nullptr);
        // k3: emits stage-4 A = bf16(x + H*k3)
        hsl_stage<false><<<grid, blk, 0, stream>>>(s3xb, s3yb, xc, yc, k2x, k2y,
                                                   0.5f * H_DT, H_DT, F, fs,
                                                   Wbf, b_hh, k3x, k3y, s4xb, s4yb,
                                                   nullptr, nullptr, nullptr, nullptr, nullptr, nullptr);
        // k4 + RK4 combine: writes f32 ping-pong state and bf16 A for next step
        hsl_stage<true><<<grid, blk, 0, stream>>>(s4xb, s4yb, xc, yc, k3x, k3y,
                                                  H_DT, 0.0f, F, fs,
                                                  Wbf, b_hh, xn, yn, xb, yb,
                                                  k1x, k1y, k2x, k2y, k3x, k3y);
        float* tx = xc; xc = xn; xn = tx;
        float* ty = yc; yc = yn; yn = ty;
    }

    hsl_out_proj<<<(B_SZ * O_SZ) / 256, 256, 0, stream>>>(xc, W_ho, b_ho, (float*)d_out);
}
